// NeuralFeatureField_37933151158467
// MI455X (gfx1250) — matrix operand order repaired and checked by reading, not compiled, not run on hardware
//
#include <hip/hip_runtime.h>
#include <hip/hip_bf16.h>

typedef _Float16 half_t;
typedef __attribute__((ext_vector_type(16))) _Float16 v16h;
typedef __attribute__((ext_vector_type(8)))  float    v8f;
typedef __attribute__((ext_vector_type(4)))  float    v4f;

#define NLVL    10
#define NFEAT   8
#define TSIZE   (1u << 20)
#define KENC    96      // 80 real features padded to 96 (multiple of WMMA K=32)
#define KREAL   80
#define HID1    384
#define HID2    768
#define ROWS    128     // rows per block
#define THREADS 256

// Scales/resolutions baked from the reference's float64 math (fp32 exp2/ceil on
// device would mis-round levels 3/6/9: e.g. 62.99999999999999 -> res 64 not 65).
__constant__ float kScale[NLVL] = {
    15.0f, 24.39840831758949f, 39.31747359663594f, 63.0f,
    100.59366732596474f, 160.26989438862868f, 255.0f,
    405.3746693034929f, 644.0795775391954f, 1023.0f };
__constant__ unsigned kRes[NLVL] = {16u,26u,41u,64u,102u,162u,256u,407u,646u,1024u};

// A fragment (16x32 f16): lane L holds row L&15; halves h map to
// K = kBase + (h/8)*16 + (L/16)*8 + (h%8)  -> two 16-byte chunks.
__device__ __forceinline__ v16h load_frag_a(const half_t* base, int rowBase,
                                            int kBase, int stride, int lane) {
    int row  = rowBase + (lane & 15);
    int koff = (lane >> 4) * 8;
    const half_t* p = base + row * stride + kBase + koff;
    union { v16h h; v4f f[2]; } u;
    u.f[0] = *(const v4f*)(p);
    u.f[1] = *(const v4f*)(p + 16);
    return u.h;
}

// B fragment (32x16 f16) from an N-major (transposed) weight matrix:
// lane L holds column L&15; halves h map to K = kBase + (h/8)*16 + (L/16)*8 + (h%8).
__device__ __forceinline__ v16h load_frag_b(const half_t* w, int colBase,
                                            int kBase, int stride, int lane) {
    int col = colBase + (lane & 15);
    int kb  = kBase + (lane >> 4) * 8;
    const half_t* p = w + col * stride + kb;
    union { v16h h; v4f f[2]; } u;
    u.f[0] = *(const v4f*)(p);
    u.f[1] = *(const v4f*)(p + 16);
    return u.h;
}

__global__ void convert_weights(const float* __restrict__ W1,
                                const float* __restrict__ W2,
                                half_t* __restrict__ W1t,
                                half_t* __restrict__ W2t) {
    int idx = blockIdx.x * blockDim.x + threadIdx.x;
    const int n1 = HID1 * KENC;     // 36864  : W1t[n][k], k padded with zeros
    const int n2 = HID2 * HID1;     // 294912 : W2t[n][k]
    if (idx < n1) {
        int n = idx / KENC, k = idx % KENC;
        W1t[idx] = (k < KREAL) ? (half_t)W1[k * HID1 + n] : (half_t)0.0f;
    } else if (idx < n1 + n2) {
        int j = idx - n1;
        int n = j / HID1, k = j % HID1;
        W2t[j] = (half_t)W2[k * HID2 + n];
    }
}

__global__ void __launch_bounds__(THREADS)
nff_fused(const float* __restrict__ coords,
          const float* __restrict__ table,
          const half_t* __restrict__ W1t,
          const float* __restrict__ bias1,
          const half_t* __restrict__ W2t,
          const float* __restrict__ bias2,
          float* __restrict__ out) {
    extern __shared__ half_t smem[];
    half_t* encT = smem;                 // ROWS x KENC  (24 KB)
    half_t* hT   = smem + ROWS * KENC;   // ROWS x HID1  (96 KB)

    const int tid   = threadIdx.x;
    const int rowT  = tid & (ROWS - 1);
    const int grp   = tid >> 7;          // 0 -> levels 0..4, 1 -> levels 5..9
    const int gRow0 = blockIdx.x * ROWS;

    // ---------------- stage 0: hash-grid encode -> LDS (f16) ----------------
    {
        const int gRow = gRow0 + rowT;
        const float cx = coords[2 * gRow + 0];
        const float cy = coords[2 * gRow + 1];
        for (int li = 0; li < 5; ++li) {
            const int lvl = grp * 5 + li;
            const float scale = kScale[lvl];
            const unsigned res = kRes[lvl];
            float px = cx * scale + 0.5f;
            float py = cy * scale + 0.5f;
            float fx = floorf(px), fy = floorf(py);
            float w1x = px - fx, w1y = py - fy;
            float w0x = 1.0f - w1x, w0y = 1.0f - w1y;
            unsigned x0 = (unsigned)fx, y0 = (unsigned)fy;
            float acc[NFEAT];
            #pragma unroll
            for (int j = 0; j < NFEAT; ++j) acc[j] = 0.0f;
            #pragma unroll
            for (int crn = 0; crn < 4; ++crn) {
                unsigned xx = x0 + (crn & 1);
                unsigned yy = y0 + (crn >> 1);
                unsigned idx = xx + yy * res;          // all levels dense (res^2 <= T)
                if (idx > TSIZE - 1u) idx = TSIZE - 1u; // edge clamp (matches jnp gather)
                float w = ((crn & 1) ? w1x : w0x) * ((crn >> 1) ? w1y : w0y);
                const float* tp = table + ((size_t)lvl * TSIZE + idx) * NFEAT;
                v4f t0 = *(const v4f*)tp;
                v4f t1 = *(const v4f*)(tp + 4);
                #pragma unroll
                for (int j = 0; j < 4; ++j) { acc[j] += w * t0[j]; acc[4 + j] += w * t1[j]; }
            }
            half_t* er = encT + rowT * KENC + lvl * NFEAT;
            #pragma unroll
            for (int j = 0; j < NFEAT; ++j) er[j] = (half_t)acc[j];
        }
        if (grp == 0) {  // zero the K padding 80..95
            half_t* er = encT + rowT * KENC + KREAL;
            #pragma unroll
            for (int j = 0; j < KENC - KREAL; ++j) er[j] = (half_t)0.0f;
        }
    }
    __syncthreads();

    const int wave   = tid >> 5;
    const int lane   = tid & 31;
    const int laneLo = lane & 15;
    const int laneHi = lane >> 4;
    const int rw     = wave * 16;        // this wave's 16-row slice of the tile

    // ---------------- stage 1: h = relu(enc @ W1 + b1) -> LDS f16 ----------------
    for (int ct4 = 0; ct4 < HID1 / 64; ++ct4) {      // 6 iters x 4 col-tiles
        v8f cacc[4] = {};
        #pragma unroll
        for (int kk = 0; kk < KENC / 32; ++kk) {     // 3 k-steps
            v16h a = load_frag_a(encT, rw, kk * 32, KENC, lane);
            #pragma unroll
            for (int f = 0; f < 4; ++f) {
                v16h b = load_frag_b(W1t, (ct4 * 4 + f) * 16, kk * 32, KENC, lane);
                cacc[f] = __builtin_amdgcn_wmma_f32_16x16x32_f16(
                    false, a, false, b, (short)0, cacc[f], false, false);
            }
        }
        #pragma unroll
        for (int f = 0; f < 4; ++f) {
            int col = (ct4 * 4 + f) * 16 + laneLo;
            float bv = bias1[col];
            #pragma unroll
            for (int v = 0; v < 8; ++v) {
                float x = cacc[f][v] + bv;
                x = fmaxf(x, 0.0f);
                hT[(rw + v + 8 * laneHi) * HID1 + col] = (half_t)x;
            }
        }
    }
    __syncthreads();

    // ---------------- stage 2: out = h @ W2 + b2 (f32) ----------------
    for (int ct4 = 0; ct4 < HID2 / 64; ++ct4) {      // 12 iters x 4 col-tiles
        v8f cacc[4] = {};
        for (int kk = 0; kk < HID1 / 32; ++kk) {     // 12 k-steps
            v16h a = load_frag_a(hT, rw, kk * 32, HID1, lane);
            #pragma unroll
            for (int f = 0; f < 4; ++f) {
                v16h b = load_frag_b(W2t, (ct4 * 4 + f) * 16, kk * 32, HID1, lane);
                cacc[f] = __builtin_amdgcn_wmma_f32_16x16x32_f16(
                    false, a, false, b, (short)0, cacc[f], false, false);
            }
        }
        #pragma unroll
        for (int f = 0; f < 4; ++f) {
            int col = (ct4 * 4 + f) * 16 + laneLo;
            float bv = bias2[col];
            #pragma unroll
            for (int v = 0; v < 8; ++v) {
                out[(size_t)(gRow0 + rw + v + 8 * laneHi) * HID2 + col] = cacc[f][v] + bv;
            }
        }
    }
}

extern "C" void kernel_launch(void* const* d_in, const int* in_sizes, int n_in,
                              void* d_out, int out_size, void* d_ws, size_t ws_size,
                              hipStream_t stream) {
    const float* coords = (const float*)d_in[0];
    const float* table  = (const float*)d_in[1];
    const float* W1     = (const float*)d_in[2];
    const float* b1     = (const float*)d_in[3];
    const float* W2     = (const float*)d_in[4];
    const float* b2     = (const float*)d_in[5];
    float* out = (float*)d_out;

    half_t* W1t = (half_t*)d_ws;                       // 384*96 f16
    half_t* W2t = W1t + (size_t)HID1 * KENC;           // 768*384 f16

    const int N = in_sizes[0] / 2;                     // 262144 points
    const int convElems = HID1 * KENC + HID2 * HID1;   // 331776
    convert_weights<<<(convElems + THREADS - 1) / THREADS, THREADS, 0, stream>>>(
        W1, W2, W1t, W2t);

    size_t smemBytes = (size_t)(ROWS * KENC + ROWS * HID1) * sizeof(half_t); // 120 KB
    nff_fused<<<N / ROWS, THREADS, smemBytes, stream>>>(
        coords, table, W1t, b1, W2t, b2, out);
}
